// TransformerEncoder_12867722019555
// MI455X (gfx1250) — compile-verified, matrix operand order checked by reading, not run on hardware
//
#include <hip/hip_runtime.h>
#include <math.h>

// ---------------------------------------------------------------------------
// Transformer encoder (B=8,S=500,D=1024,H=16,L=4,F=4096) for gfx1250 (MI455X)
// All matmuls via v_wmma_f32_16x16x32_bf16 (wave32 WMMA), fp32 accumulate.
// Activation buffers padded to M_PAD rows so all WMMA operand loads are
// unconditional (no exec-mask branches in the inner loops).
// ---------------------------------------------------------------------------

#define BB   8
#define SS   500
#define DD   1024
#define HH   16
#define LL   4
#define FF   4096
#define MM   (BB * SS)      // 4000 tokens
#define MPAD 4032           // 63 * 64, also covers attention key overrun (4011)
#define DK   64
#define SKEY 512            // padded key count (32 * 16)
#define LNEPS 1e-6f

typedef __attribute__((ext_vector_type(16))) __bf16 v16bf;
typedef __attribute__((ext_vector_type(8)))  float  v8f;

union FragU {
    v16bf v;
    uint4          q[2];
    unsigned int   u[8];
    unsigned short s[16];
};

__device__ __forceinline__ unsigned short f2bf(float f) {
    unsigned int u = __float_as_uint(f);
    u += 0x7FFFu + ((u >> 16) & 1u);          // round-to-nearest-even
    return (unsigned short)(u >> 16);
}
// K-index inside a 32-wide contraction block for 16-bit WMMA fragments:
// lanes 0-15 hold K {0..7,16..23}, lanes 16-31 hold K {8..15,24..31}
__device__ __forceinline__ int kmap(int lane, int e) {
    return (e < 8 ? e : 8 + e) + ((lane & 16) ? 8 : 0);
}
// A-fragment: two contiguous 16B chunks at element offsets 0 and 16
__device__ __forceinline__ void load_afrag(FragU& fr, const unsigned short* ap) {
    fr.q[0] = *(const uint4*)(ap);        // K 0..7   (relative)
    fr.q[1] = *(const uint4*)(ap + 16);   // K 16..23 (relative)
}

// ---------------------------------------------------------------------------
// Weight repack: fp32 W[K][N] (row-major, in->out) -> bf16 WMMA B-fragments.
// Fragment f = ntile*KT + kt ; within fragment: lane*16 + e contiguous.
// ---------------------------------------------------------------------------
__global__ void repack_w_kernel(const float* __restrict__ W,
                                unsigned short* __restrict__ out,
                                int K, int N) {
    int KT = K >> 5, NT = N >> 4;
    int tid = blockIdx.x * blockDim.x + threadIdx.x;
    if (tid >= NT * KT * 32) return;
    int f    = tid >> 5;
    int lane = tid & 31;
    int kt   = f % KT;
    int nt   = f / KT;
    int n    = nt * 16 + (lane & 15);
    FragU fr;
#pragma unroll
    for (int e = 0; e < 16; ++e) {
        int k = kt * 32 + kmap(lane, e);
        fr.s[e] = f2bf(W[(size_t)k * N + n]);
    }
    unsigned short* p = out + (size_t)f * 512 + lane * 16;
    *(uint4*)(p)     = fr.q[0];
    *(uint4*)(p + 8) = fr.q[1];
}

// ---------------------------------------------------------------------------
// Zero-fill helper (clears pad rows of q/k/v so P*V pad keys contribute 0)
// ---------------------------------------------------------------------------
__global__ void zero_u16_kernel(unsigned short* __restrict__ p, int n) {
    int i = blockIdx.x * blockDim.x + threadIdx.x;
    if (i < n) p[i] = 0;
}

// ---------------------------------------------------------------------------
// Embedding + sinusoidal positional encoding -> h fp32 [MM, DD]
// ---------------------------------------------------------------------------
__global__ void embed_pos_kernel(const int* __restrict__ x,
                                 const float* __restrict__ emb,
                                 float* __restrict__ h) {
    int t   = blockIdx.x;            // token 0..MM-1
    int s   = t % SS;
    int tok = x[t];
    const float c = -__logf(10000.0f) / (float)DD;
#pragma unroll
    for (int i = 0; i < DD / 256; ++i) {
        int d  = threadIdx.x + i * 256;
        int i2 = d & ~1;
        float ang = (float)s * __expf((float)i2 * c);
        float pe  = (d & 1) ? __cosf(ang) : __sinf(ang);
        h[(size_t)t * DD + d] = emb[(size_t)tok * DD + d] + pe;
    }
}

// ---------------------------------------------------------------------------
// LayerNorm (torch semantics: Bessel var, eps added to std).
// Writes bf16 (yb) and/or fp32 (yf).
// ---------------------------------------------------------------------------
__global__ __launch_bounds__(256)
void layernorm_kernel(const float* __restrict__ x,
                      const float* __restrict__ ga,
                      const float* __restrict__ be,
                      unsigned short* __restrict__ yb,
                      float* __restrict__ yf, int Dn) {
    __shared__ float sbuf[256];
    int row = blockIdx.x, tid = threadIdx.x;
    const float* xr = x + (size_t)row * Dn;
    float xv[4];
    float p = 0.f;
#pragma unroll
    for (int i = 0; i < 4; ++i) { xv[i] = xr[tid + i * 256]; p += xv[i]; }
    sbuf[tid] = p; __syncthreads();
    for (int s = 128; s > 0; s >>= 1) {
        if (tid < s) sbuf[tid] += sbuf[tid + s];
        __syncthreads();
    }
    float mean = sbuf[0] * (1.0f / (float)Dn);
    __syncthreads();
    p = 0.f;
#pragma unroll
    for (int i = 0; i < 4; ++i) { float d = xv[i] - mean; p += d * d; }
    sbuf[tid] = p; __syncthreads();
    for (int s = 128; s > 0; s >>= 1) {
        if (tid < s) sbuf[tid] += sbuf[tid + s];
        __syncthreads();
    }
    float var = sbuf[0] / (float)(Dn - 1);
    float inv = 1.0f / (sqrtf(var) + LNEPS);
#pragma unroll
    for (int i = 0; i < 4; ++i) {
        int d = tid + i * 256;
        float v = ga[d] * (xv[i] - mean) * inv + be[d];
        if (yb) yb[(size_t)row * Dn + d] = f2bf(v);
        if (yf) yf[(size_t)row * Dn + d] = v;
    }
}

// ---------------------------------------------------------------------------
// GEMM: C[M,N] = A[MPAD,K](bf16 row-major) * Bw(bf16 pre-swizzled)
//       [+bias][+resid][relu]. A rows >= Mtot are garbage but never stored.
// Block = 128 threads (4 waves, 2x2), wave tile 32x64, block tile 64x128.
// ---------------------------------------------------------------------------
__global__ __launch_bounds__(128)
void gemm_bf16_kernel(const unsigned short* __restrict__ A,
                      const unsigned short* __restrict__ Bw,
                      const float* __restrict__ bias,
                      const float* __restrict__ resid,
                      float* __restrict__ outF,
                      unsigned short* __restrict__ outB,
                      int Mtot, int N, int K, int doRelu) {
    int lane = threadIdx.x & 31;
    int wave = threadIdx.x >> 5;
    int l15  = lane & 15;
    int hi   = (lane >> 4) & 1;
    int rowBase = blockIdx.x * 64  + (wave & 1) * 32;
    int colBase = blockIdx.y * 128 + (wave >> 1) * 64;
    int KT = K >> 5;

    v8f acc[2][4] = {};

    const unsigned short* ap0 = A + (size_t)(rowBase + l15)      * K + hi * 8;
    const unsigned short* ap1 = A + (size_t)(rowBase + 16 + l15) * K + hi * 8;
    const unsigned short* bp0 =
        Bw + ((size_t)(colBase >> 4) * KT) * 512 + lane * 16;

    for (int kt = 0; kt < KT; ++kt) {
        FragU afr[2];
        load_afrag(afr[0], ap0 + kt * 32);
        load_afrag(afr[1], ap1 + kt * 32);
        FragU bfr[4];
#pragma unroll
        for (int nt = 0; nt < 4; ++nt) {
            const unsigned short* bp = bp0 + ((size_t)nt * KT + kt) * 512;
            __builtin_prefetch(bp + 512, 0, 1);
            bfr[nt].q[0] = *(const uint4*)(bp);
            bfr[nt].q[1] = *(const uint4*)(bp + 8);
        }
#pragma unroll
        for (int mt = 0; mt < 2; ++mt)
#pragma unroll
            for (int nt = 0; nt < 4; ++nt)
                acc[mt][nt] = __builtin_amdgcn_wmma_f32_16x16x32_bf16(
                    false, afr[mt].v, false, bfr[nt].v,
                    (short)0, acc[mt][nt], false, false);
    }

#pragma unroll
    for (int mt = 0; mt < 2; ++mt)
#pragma unroll
        for (int nt = 0; nt < 4; ++nt) {
            int n = colBase + nt * 16 + l15;
#pragma unroll
            for (int r = 0; r < 8; ++r) {
                int row = rowBase + mt * 16 + r + hi * 8;
                if (row >= Mtot) continue;
                size_t idx = (size_t)row * N + n;
                float v = acc[mt][nt][r];
                if (bias)  v += bias[n];
                if (resid) v += resid[idx];
                if (doRelu) v = fmaxf(v, 0.0f);
                if (outF) outF[idx] = v;
                if (outB) outB[idx] = f2bf(v);
            }
        }
}

// ---------------------------------------------------------------------------
// Attention: per (qtile, head, batch) block; 4 waves x 16 queries.
// scores (Q*K^T/sqrt(dk)) -> LDS fp32 -> softmax (shfl reductions) ->
// bf16 P in LDS -> P*V via WMMA -> o bf16.
// All global loads unconditional (q/k/v pad rows zeroed / masked at scores).
// Dynamic LDS: 4*16*512*4 (scores) + 4*16*512*2 (P) = 196608 bytes.
// ---------------------------------------------------------------------------
__global__ __launch_bounds__(128)
void attention_kernel(const unsigned short* __restrict__ q,
                      const unsigned short* __restrict__ k,
                      const unsigned short* __restrict__ v,
                      unsigned short* __restrict__ o) {
    extern __shared__ char smem[];
    int lane = threadIdx.x & 31;
    int wave = threadIdx.x >> 5;
    int l15  = lane & 15;
    int hi   = (lane >> 4) & 1;
    int b  = blockIdx.z;
    int hd = blockIdx.y;
    int q0 = blockIdx.x * 64 + wave * 16;           // first query of this wave

    float*          sc = (float*)smem + (size_t)wave * 16 * SKEY;
    unsigned short* pb = (unsigned short*)(smem + 4 * 16 * SKEY * 4) +
                         (size_t)wave * 16 * SKEY;

    const float inv_sqrt_dk = 0.125f;               // 1/sqrt(64)

    // ---- Phase 1: scores = Q K^T / sqrt(dk), masked beyond S ----
    const unsigned short* qp =
        q + (size_t)(b * SS + q0 + l15) * DD + hd * DK + hi * 8;
    const unsigned short* kp0 =
        k + (size_t)(b * SS + l15) * DD + hd * DK + hi * 8;

    for (int kt2 = 0; kt2 < SKEY / 16; ++kt2) {
        v8f acc = {};
        const unsigned short* kp = kp0 + (size_t)kt2 * 16 * DD;
#pragma unroll
        for (int kk = 0; kk < 2; ++kk) {
            FragU afr, bfr;
            load_afrag(afr, qp + kk * 32);
            load_afrag(bfr, kp + kk * 32);
            acc = __builtin_amdgcn_wmma_f32_16x16x32_bf16(
                false, afr.v, false, bfr.v, (short)0, acc, false, false);
        }
        int key = kt2 * 16 + l15;
#pragma unroll
        for (int r = 0; r < 8; ++r) {
            int qr = r + hi * 8;                    // query row within wave tile
            float sv = acc[r] * inv_sqrt_dk;
            if (key >= SS) sv = -1e30f;             // mask padded keys
            sc[qr * SKEY + kt2 * 16 + l15] = sv;
        }
    }
    __syncthreads();

    // ---- Phase 2: row softmax (wave-parallel, shfl reductions) ----
    for (int r = 0; r < 16; ++r) {
        float mx = -3.4e38f;
        for (int c = lane; c < SKEY; c += 32)
            mx = fmaxf(mx, sc[r * SKEY + c]);
#pragma unroll
        for (int off = 16; off > 0; off >>= 1)
            mx = fmaxf(mx, __shfl_xor(mx, off, 32));
        float sum = 0.f;
        for (int c = lane; c < SKEY; c += 32)
            sum += __expf(sc[r * SKEY + c] - mx);
#pragma unroll
        for (int off = 16; off > 0; off >>= 1)
            sum += __shfl_xor(sum, off, 32);
        float inv = 1.0f / sum;
        for (int c = lane; c < SKEY; c += 32)
            pb[r * SKEY + c] = f2bf(__expf(sc[r * SKEY + c] - mx) * inv);
    }
    __syncthreads();

    // ---- Phase 3: O = P V  (v pad rows are zeroed; P pad cols are 0) ----
#pragma unroll
    for (int nt = 0; nt < 4; ++nt) {
        v8f acc = {};
        int d = hd * DK + nt * 16 + l15;            // output head dim column
        const unsigned short* vp = v + (size_t)(b * SS) * DD + d;
        for (int kt = 0; kt < SKEY / 32; ++kt) {
            FragU afr, bfr;
            load_afrag(afr, pb + l15 * SKEY + kt * 32 + hi * 8);
#pragma unroll
            for (int e = 0; e < 16; ++e) {
                int key = kt * 32 + kmap(lane, e);
                bfr.s[e] = vp[(size_t)key * DD];
            }
            acc = __builtin_amdgcn_wmma_f32_16x16x32_bf16(
                false, afr.v, false, bfr.v, (short)0, acc, false, false);
        }
#pragma unroll
        for (int r = 0; r < 8; ++r) {
            int qr = q0 + r + hi * 8;
            if (qr < SS)
                o[(size_t)(b * SS + qr) * DD + d] = f2bf(acc[r]);
        }
    }
}

// ---------------------------------------------------------------------------
// Host-side orchestration
// ---------------------------------------------------------------------------
extern "C" void kernel_launch(void* const* d_in, const int* in_sizes, int n_in,
                              void* d_out, int out_size, void* d_ws, size_t ws_size,
                              hipStream_t stream) {
    const int*   x       = (const int*)  d_in[0];
    const float* emb     = (const float*)d_in[1];
    const float* attn_w  = (const float*)d_in[2];   // [L,4,D,D]
    const float* attn_b  = (const float*)d_in[3];   // [L,4,D]
    const float* ln_a    = (const float*)d_in[4];   // [L,2,D]
    const float* ln_b    = (const float*)d_in[5];
    const float* w1      = (const float*)d_in[6];   // [L,D,F]
    const float* b1      = (const float*)d_in[7];   // [L,F]
    const float* w2      = (const float*)d_in[8];   // [L,F,D]
    const float* b2      = (const float*)d_in[9];   // [L,D]
    const float* norm_a  = (const float*)d_in[10];
    const float* norm_b  = (const float*)d_in[11];
    float*       out     = (float*)d_out;

    char* ws = (char*)d_ws;
    size_t off = 0;
    auto take = [&](size_t bytes) { size_t o = off; off = (off + bytes + 255) & ~(size_t)255; return o; };
    const size_t szH  = (size_t)MM   * DD * 4;
    const size_t szBF = (size_t)MPAD * DD * 2;      // padded bf16 activations
    const size_t szA1 = (size_t)MPAD * FF * 2;

    float*          hbuf = (float*)         (ws + take(szH));
    unsigned short* ybuf = (unsigned short*)(ws + take(szBF));
    unsigned short* qbuf = (unsigned short*)(ws + take(szBF));
    unsigned short* kbuf = (unsigned short*)(ws + take(szBF));
    unsigned short* vbuf = (unsigned short*)(ws + take(szBF));
    unsigned short* obuf = (unsigned short*)(ws + take(szBF));
    unsigned short* a1buf= (unsigned short*)(ws + take(szA1));
    unsigned short* wAt  = (unsigned short*)(ws + take((size_t)LL * 4 * DD * DD * 2));
    unsigned short* w1s  = (unsigned short*)(ws + take((size_t)LL * DD * FF * 2));
    unsigned short* w2s  = (unsigned short*)(ws + take((size_t)LL * FF * DD * 2));
    (void)ws_size; (void)n_in; (void)in_sizes; (void)out_size;

    // ---- repack all weights to bf16 WMMA B-fragment layout ----
    auto repack = [&](const float* W, unsigned short* Wo, int K, int N) {
        int threads = (N / 16) * (K / 32) * 32;
        repack_w_kernel<<<(threads + 255) / 256, 256, 0, stream>>>(W, Wo, K, N);
    };
    for (int l = 0; l < LL; ++l) {
        for (int p = 0; p < 4; ++p)
            repack(attn_w + ((size_t)(l * 4 + p)) * DD * DD,
                   wAt    + ((size_t)(l * 4 + p)) * DD * DD, DD, DD);
        repack(w1 + (size_t)l * DD * FF, w1s + (size_t)l * DD * FF, DD, FF);
        repack(w2 + (size_t)l * FF * DD, w2s + (size_t)l * FF * DD, FF, DD);
    }

    // ---- zero pad rows of q/k/v (so P*V pad keys contribute exact zeros) ----
    {
        int padElems = (MPAD - MM) * DD;            // 32768
        int blocks   = (padElems + 255) / 256;
        zero_u16_kernel<<<blocks, 256, 0, stream>>>(qbuf + (size_t)MM * DD, padElems);
        zero_u16_kernel<<<blocks, 256, 0, stream>>>(kbuf + (size_t)MM * DD, padElems);
        zero_u16_kernel<<<blocks, 256, 0, stream>>>(vbuf + (size_t)MM * DD, padElems);
    }

    // ---- embedding + positional encoding ----
    embed_pos_kernel<<<MM, 256, 0, stream>>>(x, emb, hbuf);

    const dim3 gD(MPAD / 64, DD / 128);   // 63 x 8
    const dim3 gF(MPAD / 64, FF / 128);   // 63 x 32
    const size_t attnLds = (size_t)4 * 16 * SKEY * 4 + (size_t)4 * 16 * SKEY * 2;

    for (int l = 0; l < LL; ++l) {
        const float* la0 = ln_a + (size_t)(l * 2 + 0) * DD;
        const float* lb0 = ln_b + (size_t)(l * 2 + 0) * DD;
        const float* la1 = ln_a + (size_t)(l * 2 + 1) * DD;
        const float* lb1 = ln_b + (size_t)(l * 2 + 1) * DD;

        // --- attention sublayer ---
        layernorm_kernel<<<MM, 256, 0, stream>>>(hbuf, la0, lb0, ybuf, nullptr, DD);

        unsigned short* wq = wAt + ((size_t)(l * 4 + 0)) * DD * DD;
        unsigned short* wk = wAt + ((size_t)(l * 4 + 1)) * DD * DD;
        unsigned short* wv = wAt + ((size_t)(l * 4 + 2)) * DD * DD;
        unsigned short* wo = wAt + ((size_t)(l * 4 + 3)) * DD * DD;
        const float* bq = attn_b + (size_t)(l * 4 + 0) * DD;
        const float* bk = attn_b + (size_t)(l * 4 + 1) * DD;
        const float* bv = attn_b + (size_t)(l * 4 + 2) * DD;
        const float* bo = attn_b + (size_t)(l * 4 + 3) * DD;

        gemm_bf16_kernel<<<gD, 128, 0, stream>>>(ybuf, wq, bq, nullptr, nullptr, qbuf, MM, DD, DD, 0);
        gemm_bf16_kernel<<<gD, 128, 0, stream>>>(ybuf, wk, bk, nullptr, nullptr, kbuf, MM, DD, DD, 0);
        gemm_bf16_kernel<<<gD, 128, 0, stream>>>(ybuf, wv, bv, nullptr, nullptr, vbuf, MM, DD, DD, 0);

        attention_kernel<<<dim3(8, HH, BB), 128, attnLds, stream>>>(qbuf, kbuf, vbuf, obuf);

        // h = h + o @ Wo + bo   (fp32 residual, in place)
        gemm_bf16_kernel<<<gD, 128, 0, stream>>>(obuf, wo, bo, hbuf, hbuf, nullptr, MM, DD, DD, 0);

        // --- feed-forward sublayer ---
        layernorm_kernel<<<MM, 256, 0, stream>>>(hbuf, la1, lb1, ybuf, nullptr, DD);
        gemm_bf16_kernel<<<gF, 128, 0, stream>>>(ybuf, w1s + (size_t)l * DD * FF,
                                                 b1 + (size_t)l * FF, nullptr,
                                                 nullptr, a1buf, MM, FF, DD, 1);
        gemm_bf16_kernel<<<gD, 128, 0, stream>>>(a1buf, w2s + (size_t)l * FF * DD,
                                                 b2 + (size_t)l * DD, hbuf,
                                                 hbuf, nullptr, MM, DD, FF, 0);
    }

    // ---- final layernorm -> fp32 output ----
    layernorm_kernel<<<MM, 256, 0, stream>>>(hbuf, norm_a, norm_b, nullptr, out, DD);
}
